// DualStreamModel_90469191123014
// MI455X (gfx1250) — compile-verified, matrix operand order checked by reading, not run on hardware
//
#include <hip/hip_runtime.h>
#include <hip/hip_bf16.h>
#include <math.h>

typedef float     v8f  __attribute__((ext_vector_type(8)));
typedef _Float16  v16h __attribute__((ext_vector_type(16)));
typedef _Float16  v8h  __attribute__((ext_vector_type(8)));

#define DV 64
#define NV 200
#define K4V 152
#define KV 38
#define MSAMP 2432   // 16 * 152
#define XELEM 12800  // 64 * 200

__device__ __forceinline__ float gelu_exact(float x) {
    return 0.5f * x * (1.0f + erff(x * 0.70710678118654752f));
}

// ---------------------------------------------------------------------------
// K0: convert conv weights (5,64,64,3) f32 -> f16 (same flat layout: layer, d, c*3+t)
// ---------------------------------------------------------------------------
__global__ void wcvt_kernel(const float* __restrict__ w, _Float16* __restrict__ o, int n) {
    int i = blockIdx.x * 256 + threadIdx.x;
    if (i < n) o[i] = (_Float16)w[i];
}

// ---------------------------------------------------------------------------
// K1: gather + stem: x[m][d][n] = sum_c stem_w[d][c] * edge_data[b, e, c, n] + stem_b[d]
// ---------------------------------------------------------------------------
__global__ __launch_bounds__(256) void gather_stem_kernel(
    const float* __restrict__ edge_data, const int* __restrict__ edge_idx,
    const float* __restrict__ stem_w, const float* __restrict__ stem_b,
    float* __restrict__ xbuf)
{
    __shared__ float g[8 * NV];
    __shared__ float sw[DV * 8];
    __shared__ float sb[DV];
    const int m = blockIdx.x;
    const int b = m / K4V, j = m - b * K4V;
    const int e = edge_idx[b * K4V + j];
    const float* src = edge_data + ((size_t)b * 1560 + (size_t)e) * (8 * NV);
    __builtin_prefetch(src, 0, 1);   // global_prefetch_b8
    for (int i = threadIdx.x; i < 8 * NV; i += 256) g[i] = src[i];
    for (int i = threadIdx.x; i < DV * 8; i += 256) sw[i] = stem_w[i];
    if (threadIdx.x < DV) sb[threadIdx.x] = stem_b[threadIdx.x];
    __syncthreads();
    float* xo = xbuf + (size_t)m * XELEM;
    for (int i = threadIdx.x; i < XELEM; i += 256) {
        const int d = i / NV, n = i - d * NV;
        float s = sb[d];
#pragma unroll
        for (int c = 0; c < 8; ++c) s += sw[d * 8 + c] * g[c * NV + n];
        xo[i] = s;
    }
}

// ---------------------------------------------------------------------------
// K2: one conv block: h = conv1d(x) ; x += gelu(groupnorm(h))
// One workgroup per sample m. WMMA f16 16x16x32, f32 accumulate.
// Fragment layout fact (16-bit operands): a lane's 16 elements are two
// contiguous K-chunks {base..base+7} and {base+16..base+23}, so with a
// K-contiguous LDS layout every fragment is exactly two ds_load_b128.
// We materialize the transposed im2col  bimT[n][kk] = x[kk/3][n + kk%3 - 1]
// once per workgroup (all padding/bounds checks paid here, not per WMMA).
// Dynamic LDS layout:
//   [0,      51200)   float    hs[64][200]      conv output (f32, GN stats)
//   [51200, 131072)   _Float16 bimT[208][192]   transposed im2col (f16)
//   [131072,155648)   _Float16 wb[64][192]      weights (f16)
//   [155648,155712)   float    gmean[8], gvar[8]
// ---------------------------------------------------------------------------
__global__ __launch_bounds__(256) void conv_block_kernel(
    float* __restrict__ xbuf,
    const _Float16* __restrict__ wbf,   // [64][192] this layer, f16
    const float* __restrict__ gng, const float* __restrict__ gnb)
{
    extern __shared__ char smem[];
    float*    hs    = (float*)smem;
    _Float16* bimT  = (_Float16*)(smem + 51200);
    _Float16* wb    = (_Float16*)(smem + 131072);
    float*    gmean = (float*)(smem + 155648);
    float*    gvar  = gmean + 8;

    const int tid = threadIdx.x;
    const int m   = blockIdx.x;
    float* xg = xbuf + (size_t)m * XELEM;
    __builtin_prefetch(xg, 0, 1);

    // build transposed im2col (f16) + stage weights
    for (int i = tid; i < 192 * 208; i += 256) {
        const int kk = i / 208, n = i - kk * 208;       // coalesced in n
        const int c = kk / 3, t = kk - 3 * c;
        const int sn = n + t - 1;                        // "same" padding
        const float vv = (n < NV && sn >= 0 && sn < NV) ? xg[c * NV + sn] : 0.0f;
        bimT[n * 192 + kk] = (_Float16)vv;
    }
    for (int i = tid; i < DV * 192; i += 256) wb[i] = wbf[i];
    __syncthreads();

    const int wave = tid >> 5, lane = tid & 31;
    const int lrow = lane & 15;
    const int khi  = (lane >> 4) << 3;   // +8 K-offset for lanes 16..31

    // 52 output tiles (4 M-tiles x 13 N-tiles over N padded to 208)
    for (int tt = wave; tt < 52; tt += 8) {
        const int mt = tt / 13, nt = tt - mt * 13;
        const _Float16* arow = wb   + (mt * 16 + lrow) * 192;   // A row d
        const _Float16* brow = bimT + (nt * 16 + lrow) * 192;   // B col n
        v8f acc = {};
#pragma unroll
        for (int kt = 0; kt < 6; ++kt) {
            const int base = kt * 32 + khi;
            const v8h alo = *(const v8h*)(arow + base);
            const v8h ahi = *(const v8h*)(arow + base + 16);
            const v8h blo = *(const v8h*)(brow + base);
            const v8h bhi = *(const v8h*)(brow + base + 16);
            const v16h af = __builtin_shufflevector(alo, ahi,
                0, 1, 2, 3, 4, 5, 6, 7, 8, 9, 10, 11, 12, 13, 14, 15);
            const v16h bf = __builtin_shufflevector(blo, bhi,
                0, 1, 2, 3, 4, 5, 6, 7, 8, 9, 10, 11, 12, 13, 14, 15);
            acc = __builtin_amdgcn_wmma_f32_16x16x32_f16(
                false, af, false, bf, (short)0, acc, false, false);
        }
        // scatter D (VGPR r: lanes 0-15 -> row r, lanes 16-31 -> row r+8)
        const int col = nt * 16 + lrow;
        if (col < NV) {
            const int rbase = mt * 16 + ((lane >= 16) ? 8 : 0);
#pragma unroll
            for (int r = 0; r < 8; ++r)
                hs[(rbase + r) * NV + col] = acc[r];
        }
    }
    __syncthreads();

    // GroupNorm stats: 8 groups of 8 channels; wave w handles group w (1600 elems)
    {
        float s = 0.f, ss = 0.f;
        const float* hb = hs + wave * 8 * NV;
        for (int i = lane; i < 8 * NV; i += 32) { float v = hb[i]; s += v; ss += v * v; }
#pragma unroll
        for (int o = 16; o; o >>= 1) { s += __shfl_xor(s, o, 32); ss += __shfl_xor(ss, o, 32); }
        if (lane == 0) {
            const float mu = s * (1.0f / 1600.0f);
            gmean[wave] = mu;
            gvar[wave]  = ss * (1.0f / 1600.0f) - mu * mu;
        }
    }
    __syncthreads();

    // normalize + GELU + residual (residual re-read from global: L2-resident)
    for (int i = tid; i < XELEM; i += 256) {
        const int c = i / NV, g = c >> 3;
        const float xn = (hs[i] - gmean[g]) * rsqrtf(gvar[g] + 1e-5f) * gng[c] + gnb[c];
        xg[i] = xg[i] + gelu_exact(xn);
    }
}

// ---------------------------------------------------------------------------
// K3: conv_emb = mean_N(x) ; edge_logits = conv_emb @ eh_w.T + eh_b
// ---------------------------------------------------------------------------
__global__ __launch_bounds__(64) void emb_edge_kernel(
    const float* __restrict__ xbuf, const float* __restrict__ eh_w,
    const float* __restrict__ eh_b, float* __restrict__ conv_emb,
    float* __restrict__ edge_out)
{
    __shared__ float ce[DV];
    const int m = blockIdx.x, d = threadIdx.x;
    const float* x = xbuf + (size_t)m * XELEM + d * NV;
    float s = 0.f;
    for (int n = 0; n < NV; ++n) s += x[n];
    s *= (1.0f / (float)NV);
    conv_emb[(size_t)m * DV + d] = s;
    ce[d] = s;
    __syncthreads();
    if (d < 2) {
        float e = eh_b[d];
        for (int k = 0; k < DV; ++k) e += ce[k] * eh_w[d * DV + k];
        edge_out[(size_t)m * 2 + d] = e;
    }
}

// ---------------------------------------------------------------------------
// K4: role-merge + 2-layer transformer over S=4 tokens per node (608 nodes).
// block = 128 threads = 4 wave32; wave r owns token row r for LN reductions.
// ---------------------------------------------------------------------------
__device__ __forceinline__ void ln_row(const float (*in)[DV], float (*out)[DV],
                                       const float* gg, const float* bb,
                                       int wave, int lane, bool do_gelu)
{
    float s = 0.f, ss = 0.f;
    for (int i = lane; i < DV; i += 32) { float v = in[wave][i]; s += v; ss += v * v; }
#pragma unroll
    for (int o = 16; o; o >>= 1) { s += __shfl_xor(s, o, 32); ss += __shfl_xor(ss, o, 32); }
    const float mu = s * (1.0f / DV);
    const float rs = rsqrtf(ss * (1.0f / DV) - mu * mu + 1e-5f);
    for (int i = lane; i < DV; i += 32) {
        float xn = (in[wave][i] - mu) * rs * gg[i] + bb[i];
        out[wave][i] = do_gelu ? gelu_exact(xn) : xn;
    }
}

__global__ __launch_bounds__(128) void node_transformer_kernel(
    const float* __restrict__ conv_emb, const float* __restrict__ role_emb,
    const float* __restrict__ rm_w, const float* __restrict__ rm_b,
    const float* __restrict__ rm_ln_g, const float* __restrict__ rm_ln_b,
    const float* __restrict__ qkv_w, const float* __restrict__ qkv_b,
    const float* __restrict__ o_w, const float* __restrict__ o_b,
    const float* __restrict__ ln1_g, const float* __restrict__ ln1_b,
    const float* __restrict__ ff1_w, const float* __restrict__ ff1_b,
    const float* __restrict__ ff2_w, const float* __restrict__ ff2_b,
    const float* __restrict__ ln2_g, const float* __restrict__ ln2_b,
    float* __restrict__ node_emb)
{
    __shared__ float mM[4][DV];
    __shared__ float buf[4][DV];
    __shared__ float qkv[4][192];
    __shared__ float att[4][4][4];
    __shared__ float oa[4][DV];
    __shared__ float ffb[4][256];

    const int nk = blockIdx.x;               // b*38 + k ; conv_emb row = nk*4 + r
    const int tid = threadIdx.x, wave = tid >> 5, lane = tid & 31;
    const float* ce = conv_emb + (size_t)nk * 4 * DV;

    // merged = cat(ce4_row, role_emb_row) @ rm_w.T + rm_b
    for (int o = tid; o < 4 * DV; o += 128) {
        const int r = o >> 6, d = o & 63;
        const float* w = rm_w + d * 128;
        float s = rm_b[d];
        for (int k = 0; k < DV; ++k) s += ce[r * DV + k] * w[k];
        for (int k = 0; k < DV; ++k) s += role_emb[r * DV + k] * w[DV + k];
        buf[r][d] = s;
    }
    __syncthreads();
    ln_row(buf, mM, rm_ln_g, rm_ln_b, wave, lane, true);   // LN then GELU
    __syncthreads();

    for (int li = 0; li < 2; ++li) {
        const float* Wq = qkv_w + li * 192 * DV; const float* Bq = qkv_b + li * 192;
        for (int o = tid; o < 4 * 192; o += 128) {
            const int r = o / 192, c = o - r * 192;
            const float* w = Wq + c * DV;
            float s = Bq[c];
            for (int k = 0; k < DV; ++k) s += mM[r][k] * w[k];
            qkv[r][c] = s;
        }
        __syncthreads();
        if (tid < 64) {   // scores: (h, s, t)
            const int h = tid >> 4, sr = (tid >> 2) & 3, tc = tid & 3;
            float s = 0.f;
            for (int dq = 0; dq < 16; ++dq) s += qkv[sr][h * 16 + dq] * qkv[tc][64 + h * 16 + dq];
            att[h][sr][tc] = s * 0.25f;      // 1/sqrt(16)
        }
        __syncthreads();
        if (tid < 16) {   // softmax per (h, s)
            const int h = tid >> 2, sr = tid & 3;
            float mx = att[h][sr][0];
            for (int t = 1; t < 4; ++t) mx = fmaxf(mx, att[h][sr][t]);
            float sm = 0.f;
            for (int t = 0; t < 4; ++t) { float e = __expf(att[h][sr][t] - mx); att[h][sr][t] = e; sm += e; }
            const float inv = 1.0f / sm;
            for (int t = 0; t < 4; ++t) att[h][sr][t] *= inv;
        }
        __syncthreads();
        for (int o = tid; o < 4 * DV; o += 128) {   // o = att @ v
            const int r = o >> 6, c = o & 63, h = c >> 4;
            float s = 0.f;
            for (int t = 0; t < 4; ++t) s += att[h][r][t] * qkv[t][128 + c];
            oa[r][c] = s;
        }
        __syncthreads();
        const float* Wo = o_w + li * DV * DV; const float* Bo = o_b + li * DV;
        for (int o = tid; o < 4 * DV; o += 128) {   // proj + residual
            const int r = o >> 6, c = o & 63;
            const float* w = Wo + c * DV;
            float s = Bo[c];
            for (int k = 0; k < DV; ++k) s += oa[r][k] * w[k];
            buf[r][c] = mM[r][c] + s;
        }
        __syncthreads();
        ln_row(buf, mM, ln1_g + li * DV, ln1_b + li * DV, wave, lane, false);
        __syncthreads();
        const float* W1 = ff1_w + li * 256 * DV; const float* B1 = ff1_b + li * 256;
        for (int o = tid; o < 4 * 256; o += 128) {
            const int r = o >> 8, c = o & 255;
            const float* w = W1 + c * DV;
            float s = B1[c];
            for (int k = 0; k < DV; ++k) s += mM[r][k] * w[k];
            ffb[r][c] = gelu_exact(s);
        }
        __syncthreads();
        const float* W2 = ff2_w + li * DV * 256; const float* B2 = ff2_b + li * DV;
        for (int o = tid; o < 4 * DV; o += 128) {
            const int r = o >> 6, c = o & 63;
            const float* w = W2 + c * 256;
            float s = B2[c];
            for (int k = 0; k < 256; ++k) s += ffb[r][k] * w[k];
            buf[r][c] = mM[r][c] + s;
        }
        __syncthreads();
        ln_row(buf, mM, ln2_g + li * DV, ln2_b + li * DV, wave, lane, false);
        __syncthreads();
    }
    if (tid < DV)
        node_emb[(size_t)nk * DV + tid] =
            0.25f * (mM[0][tid] + mM[1][tid] + mM[2][tid] + mM[3][tid]);
}

// ---------------------------------------------------------------------------
// K5: ctx = gelu(LN(mean_K(ce4).reshape(4D) @ ctx_w.T + ctx_b)), per batch b
// ---------------------------------------------------------------------------
__global__ __launch_bounds__(64) void ctx_kernel(
    const float* __restrict__ conv_emb, const float* __restrict__ ctx_w,
    const float* __restrict__ ctx_b, const float* __restrict__ lng,
    const float* __restrict__ lnb, float* __restrict__ ctx)
{
    __shared__ float pools[4 * DV];
    __shared__ float cv[DV];
    __shared__ float st[2];
    const int b = blockIdx.x, tid = threadIdx.x;
    for (int o = tid; o < 4 * DV; o += 64) {
        const int r = o >> 6, d = o & 63;
        float s = 0.f;
        for (int k = 0; k < KV; ++k)
            s += conv_emb[((size_t)b * K4V + k * 4 + r) * DV + d];
        pools[o] = s * (1.0f / (float)KV);
    }
    __syncthreads();
    {
        const float* w = ctx_w + tid * 4 * DV;
        float s = ctx_b[tid];
        for (int k = 0; k < 4 * DV; ++k) s += pools[k] * w[k];
        cv[tid] = s;
    }
    __syncthreads();
    if (tid == 0) {
        float mu = 0.f;
        for (int i = 0; i < DV; ++i) mu += cv[i];
        mu *= (1.0f / DV);
        float va = 0.f;
        for (int i = 0; i < DV; ++i) { float d = cv[i] - mu; va += d * d; }
        st[0] = mu; st[1] = rsqrtf(va * (1.0f / DV) + 1e-5f);
    }
    __syncthreads();
    const float xn = (cv[tid] - st[0]) * st[1] * lng[tid] + lnb[tid];
    ctx[(size_t)b * DV + tid] = gelu_exact(xn);
}

// ---------------------------------------------------------------------------
// K6: node head: gelu(cat(node_emb, ctx) @ cls1.T + b1) @ cls2.T + b2
// ---------------------------------------------------------------------------
__global__ __launch_bounds__(64) void head_kernel(
    const float* __restrict__ node_emb, const float* __restrict__ ctx,
    const float* __restrict__ w1, const float* __restrict__ b1,
    const float* __restrict__ w2, const float* __restrict__ b2,
    float* __restrict__ out_node)
{
    __shared__ float h[DV];
    const int nk = blockIdx.x, b = nk / KV, tid = threadIdx.x;
    const float* ne = node_emb + (size_t)nk * DV;
    const float* cx = ctx + (size_t)b * DV;
    {
        const float* w = w1 + tid * 128;
        float s = b1[tid];
        for (int k = 0; k < DV; ++k) s += ne[k] * w[k];
        for (int k = 0; k < DV; ++k) s += cx[k] * w[DV + k];
        h[tid] = gelu_exact(s);
    }
    __syncthreads();
    if (tid < 8) {
        const float* w = w2 + tid * DV;
        float e = b2[tid];
        for (int k = 0; k < DV; ++k) e += h[k] * w[k];
        out_node[(size_t)nk * 8 + tid] = e;
    }
}

// ---------------------------------------------------------------------------
extern "C" void kernel_launch(void* const* d_in, const int* in_sizes, int n_in,
                              void* d_out, int out_size, void* d_ws, size_t ws_size,
                              hipStream_t stream) {
    (void)in_sizes; (void)n_in; (void)out_size; (void)ws_size;
    const float* edge_data = (const float*)d_in[0];
    const int*   edge_idx  = (const int*)d_in[1];
    // d_in[2] edge_types, d_in[3] edge_mask: unused by the reference output
    const float* stem_w   = (const float*)d_in[4];
    const float* stem_b   = (const float*)d_in[5];
    const float* conv_w   = (const float*)d_in[6];
    const float* gn_g     = (const float*)d_in[7];
    const float* gn_b     = (const float*)d_in[8];
    const float* role_emb = (const float*)d_in[9];
    const float* rm_w     = (const float*)d_in[10];
    const float* rm_b     = (const float*)d_in[11];
    const float* rm_ln_g  = (const float*)d_in[12];
    const float* rm_ln_b  = (const float*)d_in[13];
    const float* qkv_w    = (const float*)d_in[14];
    const float* qkv_b    = (const float*)d_in[15];
    const float* o_w      = (const float*)d_in[16];
    const float* o_b      = (const float*)d_in[17];
    const float* ln1_g    = (const float*)d_in[18];
    const float* ln1_b    = (const float*)d_in[19];
    const float* ff1_w    = (const float*)d_in[20];
    const float* ff1_b    = (const float*)d_in[21];
    const float* ff2_w    = (const float*)d_in[22];
    const float* ff2_b    = (const float*)d_in[23];
    const float* ln2_g    = (const float*)d_in[24];
    const float* ln2_b    = (const float*)d_in[25];
    const float* ctx_w    = (const float*)d_in[26];
    const float* ctx_b    = (const float*)d_in[27];
    const float* ctx_ln_g = (const float*)d_in[28];
    const float* ctx_ln_b = (const float*)d_in[29];
    const float* cls1_w   = (const float*)d_in[30];
    const float* cls1_b   = (const float*)d_in[31];
    const float* cls2_w   = (const float*)d_in[32];
    const float* cls2_b   = (const float*)d_in[33];
    const float* eh_w     = (const float*)d_in[34];
    const float* eh_b     = (const float*)d_in[35];

    // workspace layout (bytes)
    char* ws = (char*)d_ws;
    float*    xbuf     = (float*)ws;                         // 2432*12800*4 = 124,518,400
    float*    conv_emb = (float*)(ws + 124518400);           // 2432*64*4    =     622,592
    _Float16* wbf      = (_Float16*)(ws + 125141056);        // 5*64*192*2   =     122,880
    float*    node_emb = (float*)(ws + 125263936);           // 608*64*4     =     155,648
    float*    ctxbuf   = (float*)(ws + 125419584);           // 16*64*4      =       4,096

    float* out_node = (float*)d_out;          // (16,38,8) = 4864 floats
    float* out_edge = out_node + 4864;        // (16,152,2) = 4864 floats

    wcvt_kernel<<<(5 * 64 * 192 + 255) / 256, 256, 0, stream>>>(conv_w, wbf, 5 * 64 * 192);
    gather_stem_kernel<<<MSAMP, 256, 0, stream>>>(edge_data, edge_idx, stem_w, stem_b, xbuf);

    const size_t conv_lds = 155712;   // hs(51200) + bimT(79872) + wb(24576) + stats(64)
    for (int i = 0; i < 5; ++i) {
        conv_block_kernel<<<MSAMP, 256, conv_lds, stream>>>(
            xbuf, wbf + (size_t)i * 64 * 192, gn_g + i * DV, gn_b + i * DV);
    }

    emb_edge_kernel<<<MSAMP, 64, 0, stream>>>(xbuf, eh_w, eh_b, conv_emb, out_edge);

    node_transformer_kernel<<<608, 128, 0, stream>>>(
        conv_emb, role_emb, rm_w, rm_b, rm_ln_g, rm_ln_b,
        qkv_w, qkv_b, o_w, o_b, ln1_g, ln1_b,
        ff1_w, ff1_b, ff2_w, ff2_b, ln2_g, ln2_b, node_emb);

    ctx_kernel<<<16, 64, 0, stream>>>(conv_emb, ctx_w, ctx_b, ctx_ln_g, ctx_ln_b, ctxbuf);

    head_kernel<<<608, 64, 0, stream>>>(node_emb, ctxbuf, cls1_w, cls1_b, cls2_w, cls2_b, out_node);
}
